// WindowAttention_6004364280656
// MI455X (gfx1250) — matrix operand order repaired and checked by reading, not compiled, not run on hardware
//
#include <hip/hip_runtime.h>
#include <hip/hip_bf16.h>

// Swin-style window attention, fused: QKV proj -> QK^T -> bias+mask+softmax
// -> PV -> output proj.  One workgroup (256 thr = 8 wave32) per window-batch.
// All GEMMs via v_wmma_f32_16x16x32_bf16 (bf16 operands, fp32 accumulate).
// Weights are pre-converted to bf16 in the workspace by a tiny prep kernel.

#define NTOK   49
#define NPAD   64
#define DIMC   192
#define NHEADS 6
#define HD     32
#define QKVC   576   // 3*DIMC
#define QKC    384   // Q+K columns kept in LDS (V goes straight to Vt)
#define NWIN   64

#define QKVW_U 55296            // qkv_w bf16 size in dwords (576*192/2)
#define PRJW_U 18432            // proj_w bf16 size in dwords (192*192/2)
#define WS_NEED ((QKVW_U + PRJW_U) * 4)   // 294912 bytes

// LDS element (u16) offsets of each region
#define XBF_E  0                // 64x192 bf16 (X, later O)          24576 B
#define QK_E   12288            // 64x384 bf16 (Q | K)               49152 B
#define SB_B   (24576 + 49152)  // Sbuf byte offset: 6x64x64 f32     98304 B
#define VT_E   86016            // 6x32x64 bf16 @ byte 172032        24576 B

typedef __attribute__((ext_vector_type(16))) __bf16    v16bf;
typedef __attribute__((ext_vector_type(8)))  float     v8f;
typedef __attribute__((ext_vector_type(8)))  unsigned  v8u;

__device__ __forceinline__ unsigned short f2bf(float f) {
    __bf16 h = (__bf16)f;                    // hardware cvt, RNE
    return __builtin_bit_cast(unsigned short, h);
}

__device__ __forceinline__ unsigned pack2bf(float lo, float hi) {
    return (unsigned)f2bf(lo) | ((unsigned)f2bf(hi) << 16);
}

__device__ __forceinline__ v8f wmma_bf16(v16bf a, v16bf b, v8f c) {
    // (neg_a, A, neg_b, B, c_mod, C, reuse_a, reuse_b)
    return __builtin_amdgcn_wmma_f32_16x16x32_bf16(false, a, false, b,
                                                   (short)0, c, false, false);
}

// A-matrix fragment (16x32 bf16) from a row-major bf16 buffer (dword view).
// lane L: row = L%16; VGPR j -> K = (L<16?0:16)+2j  => 8 contiguous dwords.
__device__ __forceinline__ v16bf load_a_lds(const unsigned* base, int stride_u,
                                            int row, int k0_u, int lhalf) {
    const unsigned* p = base + row * stride_u + k0_u + lhalf * 8;
    v8u a;
    a[0] = p[0];  a[1] = p[1];  a[2] = p[2];  a[3] = p[3];
    a[4] = p[4];  a[5] = p[5];  a[6] = p[6];  a[7] = p[7];
    return __builtin_bit_cast(v16bf, a);
}

// B-matrix fragment (32x16 bf16): B[k][n] = buf[n][k] for a row-major bf16
// buffer (multiply by the buffer's transpose).  lane L: n = L%16;
// VGPR j -> K = (L<16?0:16)+2j  => 8 contiguous dwords (2x b128).
__device__ __forceinline__ v16bf load_b_rowmaj(const unsigned* base, int stride_u,
                                               int n_row, int c0_u, int lhalf) {
    const unsigned* p = base + n_row * stride_u + c0_u + lhalf * 8;
    v8u b;
#pragma unroll
    for (int j = 0; j < 8; ++j) b[j] = p[j];
    return __builtin_bit_cast(v16bf, b);
}

// Fallback B fragment from a row-major fp32 weight matrix W[n][k].
__device__ __forceinline__ v16bf load_b_w32(const float* __restrict__ W, int ldk,
                                            int n, int k0, int lhalf) {
    const float* p = W + (size_t)n * ldk + k0 + lhalf * 16;
    v8u b;
#pragma unroll
    for (int j = 0; j < 8; ++j) b[j] = pack2bf(p[2 * j], p[2 * j + 1]);
    return __builtin_bit_cast(v16bf, b);
}

// ---- prep: fp32 weights -> bf16 workspace (runs once per launch) -----------
__global__ __launch_bounds__(256)
void cvt_weights(const float* __restrict__ qkv_w, const float* __restrict__ proj_w,
                 unsigned* __restrict__ wbf) {
    int i = blockIdx.x * 256 + threadIdx.x;          // float4 index
    const float4* qv = (const float4*)qkv_w;
    const float4* pv = (const float4*)proj_w;
    if (i < (QKVC * DIMC) / 4) {
        float4 v = qv[i];
        wbf[2 * i]     = pack2bf(v.x, v.y);
        wbf[2 * i + 1] = pack2bf(v.z, v.w);
    }
    if (i < (DIMC * DIMC) / 4) {
        float4 v = pv[i];
        wbf[QKVW_U + 2 * i]     = pack2bf(v.x, v.y);
        wbf[QKVW_U + 2 * i + 1] = pack2bf(v.z, v.w);
    }
}

template <bool WBF>
__global__ __launch_bounds__(256, 1)
void swin_win_attn(const float* __restrict__ x,      // (4096,49,192)
                   const float* __restrict__ mask,   // (64,49,49)
                   const float* __restrict__ qkv_w,  // (576,192)
                   const float* __restrict__ qkv_b,  // (576)
                   const float* __restrict__ proj_w, // (192,192)
                   const float* __restrict__ proj_b, // (192)
                   const float* __restrict__ rpb,    // (169,6)
                   const int*   __restrict__ rel_idx,// (49,49)
                   const unsigned* __restrict__ wbf, // bf16 weights (or null)
                   float* __restrict__ out)          // (4096,49,192)
{
    extern __shared__ char smem[];
    unsigned short* S16  = (unsigned short*)smem;     // whole LDS as u16
    unsigned short* Xbf  = S16 + XBF_E;
    unsigned short* QK   = S16 + QK_E;
    float*          Sbuf = (float*)(smem + SB_B);
    unsigned short* Vt   = S16 + VT_E;

    const int tid   = threadIdx.x;
    // readfirstlane: tell the compiler wave id (and everything derived from
    // it: tile ids, region selects) is scalar -> SALU addressing, s_cselect
    // instead of EXEC-mask branching in the store epilogues.
    const int wave  = __builtin_amdgcn_readfirstlane(tid >> 5);
    const int lane  = tid & 31;
    const int l16   = lane & 15;
    const int lhalf = lane >> 4;
    const int blk   = blockIdx.x;
    const int win   = blk & (NWIN - 1);
    const float scale = 0.17677669529663689f;  // HEAD_DIM^-0.5

    // ---- phase 0: X -> bf16 in LDS, zero-pad rows 49..63 -------------------
    {
        const float4* xv = (const float4*)(x + (size_t)blk * NTOK * DIMC);
        uint2* Xv = (uint2*)Xbf;
        for (int i = tid; i < (NPAD * DIMC) / 4; i += 256) {
            int row = (i * 4) / DIMC;                // 192 % 4 == 0
            float4 v = (row < NTOK) ? xv[i] : float4{0.f, 0.f, 0.f, 0.f};
            Xv[i] = uint2{pack2bf(v.x, v.y), pack2bf(v.z, v.w)};
        }
    }
    __syncthreads();

    // ---- phase 1: QKV = X @ qkv_w^T + qkv_b  (Q pre-scaled) ----------------
    // Q,K tiles -> QK buffer; V tiles go straight into the token-major Vt
    // buffer.  Destination = scalar base offset + scalar stride (no branch).
    const unsigned* Xu = (const unsigned*)Xbf;   // 96 dwords per row
#pragma unroll 1
    for (int t = wave; t < 144; t += 8) {        // 4 (M) x 36 (N) tiles
        int mt = t / 36, nt = t - mt * 36;
        int m0 = mt * 16, n0 = nt * 16;
        int arow = m0 + l16;
        int bcol = n0 + l16;
        v8f acc = {};
#pragma unroll 1
        for (int kk = 0; kk < 6; ++kk) {
            v16bf a = load_a_lds(Xu, 96, arow, kk * 16, lhalf);
            v16bf bm;
            if constexpr (WBF) bm = load_b_rowmaj(wbf, 96, bcol, kk * 16, lhalf);
            else               bm = load_b_w32(qkv_w, DIMC, bcol, kk * 32, lhalf);
            acc = wmma_bf16(a, bm, acc);
        }
        float bias = qkv_b[bcol];
        float sc   = (n0 < DIMC) ? scale : 1.0f;   // scale only Q columns
        bool  isV  = (n0 >= 2 * DIMC);
        int   row0 = m0 + lhalf * 8;
        // element offsets in S16 for r = 0 (both arms always valid indices)
        int offV = VT_E + ((n0 - 2 * DIMC >= 0 ? n0 - 2 * DIMC : 0) >> 5) * 2048
                        + ((bcol - 2 * DIMC) & 31) * 64 + row0;
        int offQ = QK_E + row0 * QKC + bcol;
        int off0 = isV ? offV : offQ;
        int step = isV ? 1 : QKC;                  // per-r element stride
#pragma unroll
        for (int r = 0; r < 8; ++r)
            S16[off0 + r * step] = f2bf((acc[r] + bias) * sc);
    }
    __syncthreads();

    // ---- phase 2: S_h = Q_h @ K_h^T  (one WMMA per tile, K-dim = 32) -------
    const unsigned* Qu = (const unsigned*)QK;    // 192 dwords per row
#pragma unroll 1
    for (int t = wave; t < 96; t += 8) {         // 6 heads x (4x4) tiles
        int h  = t / 16, r = t - h * 16;
        int mt = r >> 2, nt = r & 3;
        int m0 = mt * 16, n0 = nt * 16;
        v16bf a  = load_a_lds(Qu, 192, m0 + l16, h * 16, lhalf);          // Q cols h*32..
        v16bf bm = load_b_rowmaj(Qu, 192, n0 + l16, 96 + h * 16, lhalf);  // K cols 192+h*32..
        v8f acc = {};
        acc = wmma_bf16(a, bm, acc);
        float* Sh = Sbuf + h * 4096 + (m0 + lhalf * 8) * 64 + n0 + l16;
#pragma unroll
        for (int rr = 0; rr < 8; ++rr) Sh[rr * 64] = acc[rr];
    }
    __syncthreads();

    // ---- phase 3: bias + mask + online softmax; write P as bf16 in place ---
#pragma unroll 1
    for (int t = tid; t < NHEADS * NTOK; t += 256) {
        int h = t / NTOK, i = t - h * NTOK;
        float* Srow = Sbuf + h * 4096 + i * 64;
        const int*   ridx = rel_idx + i * NTOK;
        const float* mrow = mask + (size_t)win * (NTOK * NTOK) + i * NTOK;
        float mx = -3.0e38f, sum = 0.0f;
        for (int j = 0; j < NTOK; ++j) {
            float v = Srow[j] + rpb[ridx[j] * NHEADS + h] + mrow[j];
            if (v > mx) { sum *= __expf(mx - v); mx = v; }
            sum += __expf(v - mx);
        }
        float rinv = 1.0f / sum;
        unsigned short* Prow = (unsigned short*)Srow;  // forward overwrite is safe
        for (int j = 0; j < NTOK; ++j) {
            float v = Srow[j] + rpb[ridx[j] * NHEADS + h] + mrow[j];
            Prow[j] = f2bf(__expf(v - mx) * rinv);
        }
        for (int j = NTOK; j < 64; ++j) Prow[j] = 0;   // zero pad cols
    }
    __syncthreads();

    // ---- phase 4: O_h = P_h @ V_h  (K-dim = 64 padded tokens) --------------
    const unsigned* Pu  = (const unsigned*)Sbuf;  // per head: 64 rows x 64 dwords
    const unsigned* Vtu = (const unsigned*)Vt;    // per head: 32 rows x 32 dwords
#pragma unroll 1
    for (int t = wave; t < 48; t += 8) {          // 6 heads x (4x2) tiles
        int h  = t / 8, r = t - h * 8;
        int mt = r >> 1, n2 = r & 1;
        int m0 = mt * 16;
        v8f acc = {};
#pragma unroll 1
        for (int kk = 0; kk < 2; ++kk) {
            v16bf a  = load_a_lds(Pu + h * 4096, 64, m0 + l16, kk * 16, lhalf);
            // B[k][n] = Vt[h][d = n2*16 + lane][tok = k]: contiguous token pairs
            v16bf bm = load_b_rowmaj(Vtu + h * 1024, 32, n2 * 16 + l16,
                                     kk * 16, lhalf);
            acc = wmma_bf16(a, bm, acc);
        }
        unsigned short* Od = Xbf + (m0 + lhalf * 8) * DIMC
                                 + h * 32 + n2 * 16 + l16;
#pragma unroll
        for (int rr = 0; rr < 8; ++rr) Od[rr * DIMC] = f2bf(acc[rr]);
    }
    __syncthreads();

    // ---- phase 5: out = O @ proj_w^T + proj_b ------------------------------
    const unsigned* Ou = (const unsigned*)Xbf;
    float* ob = out + (size_t)blk * NTOK * DIMC;
#pragma unroll 1
    for (int t = wave; t < 48; t += 8) {         // 4 (M) x 12 (N) tiles
        int mt = t / 12, nt = t - mt * 12;
        int m0 = mt * 16, n0 = nt * 16;
        int arow = m0 + l16;
        int bcol = n0 + l16;
        v8f acc = {};
#pragma unroll 1
        for (int kk = 0; kk < 6; ++kk) {
            v16bf a = load_a_lds(Ou, 96, arow, kk * 16, lhalf);
            v16bf bm;
            if constexpr (WBF) bm = load_b_rowmaj(wbf + QKVW_U, 96, bcol, kk * 16, lhalf);
            else               bm = load_b_w32(proj_w, DIMC, bcol, kk * 32, lhalf);
            acc = wmma_bf16(a, bm, acc);
        }
        float pb = proj_b[bcol];
#pragma unroll
        for (int rr = 0; rr < 8; ++rr) {
            int row = m0 + rr + lhalf * 8;
            if (row < NTOK) ob[row * DIMC + bcol] = acc[rr] + pb;
        }
    }
}

extern "C" void kernel_launch(void* const* d_in, const int* in_sizes, int n_in,
                              void* d_out, int out_size, void* d_ws, size_t ws_size,
                              hipStream_t stream) {
    const float* x      = (const float*)d_in[0];
    const float* mask   = (const float*)d_in[1];
    const float* qkv_w  = (const float*)d_in[2];
    const float* qkv_b  = (const float*)d_in[3];
    const float* proj_w = (const float*)d_in[4];
    const float* proj_b = (const float*)d_in[5];
    const float* rpb    = (const float*)d_in[6];
    const int*   ridx   = (const int*)d_in[7];
    float* out = (float*)d_out;

    const size_t smem = 24576 + 49152 + 98304 + 24576;  // 192 KB (WGP has 320 KB)
    const bool wbf_ok = (ws_size >= (size_t)WS_NEED);

    if (wbf_ok) {
        unsigned* wbf = (unsigned*)d_ws;
        int nb = ((QKVC * DIMC) / 4 + 255) / 256;       // covers proj_w too
        hipLaunchKernelGGL(cvt_weights, dim3(nb), dim3(256), 0, stream,
                           qkv_w, proj_w, wbf);
        hipFuncSetAttribute((const void*)&swin_win_attn<true>,
                            hipFuncAttributeMaxDynamicSharedMemorySize, (int)smem);
        hipLaunchKernelGGL((swin_win_attn<true>), dim3(4096), dim3(256), smem, stream,
                           x, mask, qkv_w, qkv_b, proj_w, proj_b, rpb, ridx,
                           (const unsigned*)wbf, out);
    } else {
        hipFuncSetAttribute((const void*)&swin_win_attn<false>,
                            hipFuncAttributeMaxDynamicSharedMemorySize, (int)smem);
        hipLaunchKernelGGL((swin_win_attn<false>), dim3(4096), dim3(256), smem, stream,
                           x, mask, qkv_w, qkv_b, proj_w, proj_b, rpb, ridx,
                           (const unsigned*)nullptr, out);
    }
}
